// MultiHeadAttention_4758823764165
// MI455X (gfx1250) — matrix operand order repaired and checked by reading, not compiled, not run on hardware
//
#include <hip/hip_runtime.h>
#include <hip/hip_bf16.h>

typedef __attribute__((ext_vector_type(16))) __bf16 v16bf;
typedef __attribute__((ext_vector_type(8)))  __bf16 v8bf;
typedef __attribute__((ext_vector_type(8)))  float  v8f;

#define BATCH 128
#define NBLK  64
#define DMODEL 512
#define NHEAD 8
#define HDIM  64
#define TOPK  16
#define EPSV  1e-8f

// round-to-nearest-even f32 -> bf16 (bit manipulation)
__device__ __forceinline__ unsigned short f2bf(float f) {
  union { float f; unsigned int u; } c; c.f = f;
  unsigned int u = c.u;
  return (unsigned short)((u + 0x7FFFu + ((u >> 16) & 1u)) >> 16);
}

// A operand: 16x32 bf16 tile, lane(r=lane&15)=row, half h=lane>>4.
// elems 0..7 -> K = 8h..8h+7 ; elems 8..15 -> K = 16+8h..16+8h+7
__device__ __forceinline__ v16bf frag_a(const unsigned short* base, int stride) {
  const int lane = threadIdx.x & 31;
  const unsigned short* p = base + (lane & 15) * stride + 8 * (lane >> 4);
  v8bf lo = *(const v8bf*)(p);
  v8bf hi = *(const v8bf*)(p + 16);
  return __builtin_shufflevector(lo, hi, 0,1,2,3,4,5,6,7,8,9,10,11,12,13,14,15);
}

// B operand from B^T tile (N x K layout): lane(n=lane&15)=column N, elems 0..7 -> K = 8h+i ; elems 8..15 -> K = 16+8h+(i-8)
__device__ __forceinline__ v16bf frag_b(const unsigned short* base, int stride) {
  const int lane = threadIdx.x & 31;
  const unsigned short* p = base + (lane & 15) * stride + 8 * (lane >> 4);
  v8bf lo = *(const v8bf*)(p);
  v8bf hi = *(const v8bf*)(p + 16);
  return __builtin_shufflevector(lo, hi, 0,1,2,3,4,5,6,7,8,9,10,11,12,13,14,15);
}

__device__ __forceinline__ v8f wmma_bf16(v16bf a, v16bf b, v8f c) {
  return __builtin_amdgcn_wmma_f32_16x16x32_bf16(false, a, false, b, (short)0, c, false, false);
}

// ---------------------------------------------------------------------------
// Stage 1: grouped projection.  Per block n: Y = X[:,n,:] (128x512) @ W[n] (512x512)
// grid = (8 Ntiles, 64 blocks), 256 threads (8 waves), 128x64 tile per WG.
// Full-M tile => each W element read from HBM exactly once (weights dominate bytes).
// TRANSPOSED=false: Y[b][h][n][dk]  (Q,K) ; true: Y[b][h][dk][n]  (V, pre-transposed)
// ---------------------------------------------------------------------------
template <bool TRANSPOSED>
__global__ __launch_bounds__(256)
void proj_kernel(const float* __restrict__ X, const float* __restrict__ W,
                 unsigned short* __restrict__ Y) {
  __shared__ __attribute__((aligned(16))) unsigned short As[128 * 40];
  __shared__ __attribute__((aligned(16))) unsigned short Bt[64 * 40];
  const int t = threadIdx.x;          // 0..255
  const int wave = t >> 5;            // 0..7
  const int blockn = blockIdx.y;
  const int n0 = blockIdx.x * 64;
  const int wm = (wave & 3) * 32;     // 4 M-quadrants of 128
  const int wn = (wave >> 2) * 32;    // 2 N-quadrants of 64

  v8f acc[2][2] = {};
  for (int kk = 0; kk < DMODEL; kk += 32) {
    // A tile 128x32 (rows = full batch), f32 -> bf16
    #pragma unroll
    for (int i = 0; i < 16; ++i) {
      int lin = i * 256 + t;
      int row = lin >> 5, col = lin & 31;
      As[row * 40 + col] =
          f2bf(X[(size_t)row * (NBLK * DMODEL) + (size_t)blockn * DMODEL + kk + col]);
    }
    // B tile 32x64 -> transposed (N x K) bf16
    #pragma unroll
    for (int i = 0; i < 8; ++i) {
      int lin = i * 256 + t;
      int kr = lin >> 6, oc = lin & 63;
      Bt[oc * 40 + kr] =
          f2bf(W[(size_t)blockn * (DMODEL * DMODEL) + (size_t)(kk + kr) * DMODEL + n0 + oc]);
    }
    __syncthreads();
    v16bf a0 = frag_a(&As[(wm)      * 40], 40);
    v16bf a1 = frag_a(&As[(wm + 16) * 40], 40);
    v16bf b0 = frag_b(&Bt[(wn)      * 40], 40);
    v16bf b1 = frag_b(&Bt[(wn + 16) * 40], 40);
    acc[0][0] = wmma_bf16(a0, b0, acc[0][0]);
    acc[0][1] = wmma_bf16(a0, b1, acc[0][1]);
    acc[1][0] = wmma_bf16(a1, b0, acc[1][0]);
    acc[1][1] = wmma_bf16(a1, b1, acc[1][1]);
    __syncthreads();
  }
  const int lane = t & 31;
  const int cn = lane & 15;
  const int rh = (lane >> 4) * 8;
  #pragma unroll
  for (int i = 0; i < 2; ++i)
    #pragma unroll
    for (int j = 0; j < 2; ++j)
      #pragma unroll
      for (int e = 0; e < 8; ++e) {
        int bb = wm + 16 * i + rh + e;           // batch index
        int o  = n0 + wn + 16 * j + cn;
        int h  = o >> 6, dk = o & 63;
        size_t idx;
        if (TRANSPOSED)
          idx = ((size_t)(bb * NHEAD + h) * HDIM + dk) * NBLK + blockn;
        else
          idx = ((size_t)(bb * NHEAD + h) * NBLK + blockn) * HDIM + dk;
        Y[idx] = f2bf(acc[i][j][e]);
      }
}

// ---------------------------------------------------------------------------
// Stage 2: attention per (b,h).  grid = 1024, 128 threads (4 waves).
// V tile (8KB bf16) prefetched to LDS via global_load_async_to_lds_b128, overlapped
// with QK^T + softmax; waited with s_wait_asynccnt before the PV GEMM.
// ---------------------------------------------------------------------------
__global__ __launch_bounds__(128)
void attn_kernel(const unsigned short* __restrict__ qh,
                 const unsigned short* __restrict__ kh,
                 const unsigned short* __restrict__ vt,
                 unsigned short* __restrict__ outws,
                 float* __restrict__ attn_out) {
  __shared__ __attribute__((aligned(16))) float S[64 * 64];
  __shared__ __attribute__((aligned(16))) unsigned short Asp[64 * 72];
  __shared__ __attribute__((aligned(16))) unsigned short Vs[64 * 64];
  const int t = threadIdx.x;
  const int wave = t >> 5;
  const int lane = t & 31;
  const int bh = blockIdx.x;
  const int b = bh >> 3, h = bh & 7;
  const unsigned short* Q = qh + (size_t)bh * 4096;
  const unsigned short* K = kh + (size_t)bh * 4096;
  const unsigned short* V = vt + (size_t)bh * 4096;

  // --- async prefetch of V tile (64x64 bf16 = 8192B): 128 lanes x 16B x 4 rounds ---
  {
    unsigned ldsbase = (unsigned)(unsigned long long)(const void*)&Vs[0];
    unsigned long long vbase = (unsigned long long)(const void*)V;
    #pragma unroll
    for (int r = 0; r < 4; ++r) {
      unsigned l = ldsbase + (unsigned)(t * 16 + r * 2048);
      unsigned g = (unsigned)(t * 16 + r * 2048);
      asm volatile("global_load_async_to_lds_b128 %0, %1, %2"
                   :: "v"(l), "v"(g), "s"(vbase) : "memory");
    }
  }

  // scores: wave computes rows wave*16..+15, all 64 cols. B^T of K^T is K itself.
  v8f acc[4] = {};
  #pragma unroll
  for (int ks = 0; ks < 64; ks += 32) {
    v16bf a = frag_a(Q + (wave * 16) * 64 + ks, 64);
    #pragma unroll
    for (int j = 0; j < 4; ++j) {
      v16bf bfr = frag_b(K + (j * 16) * 64 + ks, 64);
      acc[j] = wmma_bf16(a, bfr, acc[j]);
    }
  }
  const int cn = lane & 15;
  const int rh = (lane >> 4) * 8;
  const float scale = 0.125f;  // 1/sqrt(dk)
  #pragma unroll
  for (int j = 0; j < 4; ++j)
    #pragma unroll
    for (int e = 0; e < 8; ++e)
      S[(wave * 16 + rh + e) * 64 + (j * 16 + cn)] = acc[j][e] * scale;
  __syncthreads();

  // softmax + exact top-16 sparsification (one thread per query row)
  if (t < 64) {
    float* r = &S[t * 64];
    float mx = -3.4e38f;
    for (int i = 0; i < 64; ++i) mx = fmaxf(mx, r[i]);
    float sum = 0.f;
    for (int i = 0; i < 64; ++i) { float e = __expf(r[i] - mx); r[i] = e; sum += e; }
    // 16th-largest via tie-aware descending max passes (monotone in e-domain)
    float cur = 3.4e38f; int removed = 0;
    while (removed < TOPK) {
      float m = -1.f;
      for (int i = 0; i < 64; ++i) { float e = r[i]; if (e < cur && e > m) m = e; }
      if (m < 0.f) break;
      int c = 0;
      for (int i = 0; i < 64; ++i) c += (r[i] == m) ? 1 : 0;
      removed += c; cur = m;
    }
    const float inv_sum = 1.0f / sum;
    const float delta = cur * inv_sum + EPSV;   // k-th largest attn + eps
    float sumw = 0.f;
    for (int i = 0; i < 64; ++i) sumw += fmaxf(r[i] * inv_sum - delta, 0.f);
    const float invw = 1.0f / (sumw + EPSV);
    float* ao = attn_out + ((size_t)bh * 64 + t) * 64;
    for (int i = 0; i < 64; ++i) {
      float w = fmaxf(r[i] * inv_sum - delta, 0.f) * invw;
      ao[i] = w;                      // second reference output: attn_sp (b,h,q,k)
      Asp[t * 72 + i] = f2bf(w);
    }
  }
  // V prefetch must be complete and visible to all waves before the PV GEMM
  asm volatile("s_wait_asynccnt 0" ::: "memory");
  __syncthreads();

  // out = attn_sp (64x64) @ V ; B^T = V^T staged in LDS (dv-major)
  v8f acc2[4] = {};
  #pragma unroll
  for (int ks = 0; ks < 64; ks += 32) {
    v16bf a = frag_a(&Asp[(wave * 16) * 72 + ks], 72);
    #pragma unroll
    for (int j = 0; j < 4; ++j) {
      v16bf bfr = frag_b(&Vs[(j * 16) * 64 + ks], 64);
      acc2[j] = wmma_bf16(a, bfr, acc2[j]);
    }
  }
  #pragma unroll
  for (int j = 0; j < 4; ++j)
    #pragma unroll
    for (int e = 0; e < 8; ++e) {
      int qrow = wave * 16 + rh + e;
      int dv = j * 16 + cn;
      outws[(size_t)(b * NBLK + qrow) * DMODEL + h * HDIM + dv] = f2bf(acc2[j][e]);
    }
}

// ---------------------------------------------------------------------------
// Stage 3: fused fc + gate GEMMs (8192 x 512 x 512 each) + sigmoid/tanh epilogue
// grid = (8 Ntiles, 64 Mtiles), 256 threads (8 waves), 128x64 tile for both GEMMs.
// ---------------------------------------------------------------------------
__global__ __launch_bounds__(256)
void fcgate_kernel(const unsigned short* __restrict__ A,
                   const float* __restrict__ Wfc, const float* __restrict__ bfc,
                   const float* __restrict__ Wg,  const float* __restrict__ bg,
                   float* __restrict__ out) {
  __shared__ __attribute__((aligned(16))) unsigned short Bt1[64 * 40];
  __shared__ __attribute__((aligned(16))) unsigned short Bt2[64 * 40];
  const int t = threadIdx.x;          // 0..255
  const int wave = t >> 5;            // 0..7
  const int m0 = blockIdx.y * 128;
  const int n0 = blockIdx.x * 64;
  const int wm = (wave & 3) * 32;
  const int wn = (wave >> 2) * 32;

  v8f afc[2][2] = {};
  v8f ag[2][2] = {};
  for (int kk = 0; kk < DMODEL; kk += 32) {
    #pragma unroll
    for (int i = 0; i < 8; ++i) {
      int lin = i * 256 + t;
      int kr = lin >> 6, oc = lin & 63;
      size_t widx = (size_t)(kk + kr) * DMODEL + n0 + oc;
      Bt1[oc * 40 + kr] = f2bf(Wfc[widx]);
      Bt2[oc * 40 + kr] = f2bf(Wg[widx]);
    }
    __syncthreads();
    v16bf a0  = frag_a(A + (size_t)(m0 + wm)      * DMODEL + kk, DMODEL);
    v16bf a1  = frag_a(A + (size_t)(m0 + wm + 16) * DMODEL + kk, DMODEL);
    v16bf b10 = frag_b(&Bt1[(wn)      * 40], 40);
    v16bf b11 = frag_b(&Bt1[(wn + 16) * 40], 40);
    v16bf b20 = frag_b(&Bt2[(wn)      * 40], 40);
    v16bf b21 = frag_b(&Bt2[(wn + 16) * 40], 40);
    afc[0][0] = wmma_bf16(a0, b10, afc[0][0]);
    afc[0][1] = wmma_bf16(a0, b11, afc[0][1]);
    afc[1][0] = wmma_bf16(a1, b10, afc[1][0]);
    afc[1][1] = wmma_bf16(a1, b11, afc[1][1]);
    ag[0][0]  = wmma_bf16(a0, b20, ag[0][0]);
    ag[0][1]  = wmma_bf16(a0, b21, ag[0][1]);
    ag[1][0]  = wmma_bf16(a1, b20, ag[1][0]);
    ag[1][1]  = wmma_bf16(a1, b21, ag[1][1]);
    __syncthreads();
  }
  const int lane = t & 31;
  const int cn = lane & 15;
  const int rh = (lane >> 4) * 8;
  #pragma unroll
  for (int i = 0; i < 2; ++i)
    #pragma unroll
    for (int j = 0; j < 2; ++j)
      #pragma unroll
      for (int e = 0; e < 8; ++e) {
        int gm = m0 + wm + 16 * i + rh + e;
        int o  = n0 + wn + 16 * j + cn;
        float fv = afc[i][j][e] + bfc[o];
        float gv = ag[i][j][e]  + bg[o];
        float s  = 1.0f / (1.0f + __expf(-gv));
        out[(size_t)gm * DMODEL + o] = s * tanhf(fv);
      }
}

// ---------------------------------------------------------------------------
extern "C" void kernel_launch(void* const* d_in, const int* in_sizes, int n_in,
                              void* d_out, int out_size, void* d_ws, size_t ws_size,
                              hipStream_t stream) {
  const float* q   = (const float*)d_in[0];
  const float* k   = (const float*)d_in[1];
  const float* v   = (const float*)d_in[2];
  const float* Wq  = (const float*)d_in[3];
  const float* Wk  = (const float*)d_in[4];
  const float* Wv  = (const float*)d_in[5];
  const float* Wfc = (const float*)d_in[6];
  const float* bfc = (const float*)d_in[7];
  const float* Wg  = (const float*)d_in[8];
  const float* bg  = (const float*)d_in[9];

  float* out = (float*)d_out;                                  // (B, NB, DOUT)
  float* attn_out = out + (size_t)BATCH * NBLK * DMODEL;       // (B, H, NB, NB)

  unsigned short* ws = (unsigned short*)d_ws;
  const size_t seg = (size_t)BATCH * NHEAD * NBLK * HDIM;      // 4,194,304 bf16 elems
  unsigned short* qh_ws  = ws;
  unsigned short* kh_ws  = ws + seg;
  unsigned short* vt_ws  = ws + 2 * seg;
  unsigned short* out_ws = ws + 3 * seg;

  dim3 g1(8, 64);
  proj_kernel<false><<<g1, dim3(256), 0, stream>>>(q, Wq, qh_ws);
  proj_kernel<false><<<g1, dim3(256), 0, stream>>>(k, Wk, kh_ws);
  proj_kernel<true ><<<g1, dim3(256), 0, stream>>>(v, Wv, vt_ws);
  attn_kernel<<<dim3(BATCH * NHEAD), dim3(128), 0, stream>>>(qh_ws, kh_ws, vt_ws, out_ws, attn_out);
  fcgate_kernel<<<dim3(8, 64), dim3(256), 0, stream>>>(out_ws, Wfc, bfc, Wg, bg, out);
}
